// GNN_18820546691351
// MI455X (gfx1250) — compile-verified, matrix operand order checked by reading, not run on hardware
//
#include <hip/hip_runtime.h>
#include <hip/hip_bf16.h>

typedef __attribute__((ext_vector_type(16))) _Float16 v16h;
typedef __attribute__((ext_vector_type(8)))  float    v8f;

#define NN 768
#define EDGES_PER (NN - 1)          // 767 edges per sender
#define CHUNKS 48                   // 48*16 = 768 slots (last one padded)
#define CUTOFF_INV 0.1f             // 1/10.0
#define RBF_SCALE 0.44721359549995793f  // sqrt(2/10)

// silu via hardware v_rcp_f32 (one trans op instead of IEEE div sequence)
__device__ __forceinline__ float silu_f(float x) {
    return x * __builtin_amdgcn_rcpf(1.0f + __expf(-x));
}

// ---------------------------------------------------------------------------
// K1: per-node precompute for layer 1.
//   emb0[i] = embed_table[charges[i]]           (32)
//   A1[i]   = emb0[i] @ mp1_w0[0:32,:]  + b0    (32)   (sender part, bias folded)
//   B1pk[i] = emb0[i] @ mp1_w0[32:64,:]         (32)   pair-interleaved:
//             B1pk[i*32 + 2*(n%16) + n/16] = col n   -> float2 (n, n+16)
// Also zeroes the 160-wide global agg accumulator.
// ---------------------------------------------------------------------------
__global__ __launch_bounds__(256) void node_init_kernel(
    const int* __restrict__ charges, const float* __restrict__ embT,
    const float* __restrict__ mp1_w0, const float* __restrict__ mp1_b0,
    float* __restrict__ emb0, float* __restrict__ A1, float* __restrict__ B1pk,
    float* __restrict__ aggsum)
{
    if (blockIdx.x == 0 && threadIdx.x < 160) aggsum[threadIdx.x] = 0.0f;
    int t = blockIdx.x * 256 + threadIdx.x;       // 96 blocks * 256 = 768*32
    int i = t >> 5, n = t & 31;
    if (i >= NN) return;
    int q = charges[i];
    float a = mp1_b0[n], b = 0.0f;
    #pragma unroll
    for (int k = 0; k < 32; ++k) {
        float e = embT[q * 32 + k];
        a += e * mp1_w0[k * 32 + n];
        b += e * mp1_w0[(32 + k) * 32 + n];
    }
    emb0[i * 32 + n] = embT[q * 32 + n];
    A1[i * 32 + n] = a;
    B1pk[i * 32 + ((n & 15) << 1) + (n >> 4)] = b;
}

// ---------------------------------------------------------------------------
// K2/K4: per-sender edge message pass (WMMA).
//   msg[s] = (1/767) * ( sum_e silu( A[s] + B[r_e] + rbf_e @ Wrbf ) ) @ W1 + b1
// One workgroup (8 waves) per sender; each wave owns 6 chunks of 16 edges.
// rbf_e @ Wrbf done with V_WMMA_F32_16X16X32_F16 (two 16-col output tiles).
// A-layout: lane L in [0,16): row M=L, K = {0..7,16..23}; lane L+16: K = {8..15,24..31}.
// C-layout: VGPR g -> M = g (lanes 0-15) / g+8 (lanes 16-31), N = lane%16 (+16 tile1).
// Receiver of edge (s, ei): eg = s*767+ei = s*768+(ei-s) -> r = ei + (ei >= s).
// B is stored pair-interleaved so (col n, col n+16) is one b64 gather.
// Only the last chunk (47) contains the pad edge; its body is a specialized
// masked version (wave-uniform branch, EXEC all-ones around WMMA).
// ---------------------------------------------------------------------------
__global__ __launch_bounds__(256) void edge_msg_kernel(
    const float* __restrict__ nuclei, const float* __restrict__ fparam,
    const float* __restrict__ Anode,  const float2* __restrict__ Bpk,
    const float* __restrict__ Wrbf,   const float* __restrict__ W1,
    const float* __restrict__ b1,     float* __restrict__ msg)
{
    const int s    = blockIdx.x;
    const int tid  = threadIdx.x;
    const int lane = tid & 31;
    const int wave = tid >> 5;
    const int n    = lane & 15;   // column (tile0) / row-of-A this lane serves
    const int kh   = lane >> 4;   // which K-half this lane holds

    __shared__ float redbuf[8][32];
    __shared__ float hsumS[32];

    // K indices handled by this lane (identical mapping assumed for A rows / B cols)
    int   kidx[16];
    float fk[16];
    #pragma unroll
    for (int e = 0; e < 16; ++e) {
        int K = (e < 8) ? (kh * 8 + e) : (16 + kh * 8 + (e - 8));
        kidx[e] = K;
        fk[e] = fparam[K] * CUTOFF_INV;
    }

    // Preload the two f16 B tiles of Wrbf (cols n and n+16)
    v16h wb0, wb1;
    #pragma unroll
    for (int e = 0; e < 16; ++e) {
        wb0[e] = (_Float16)Wrbf[kidx[e] * 32 + n];
        wb1[e] = (_Float16)Wrbf[kidx[e] * 32 + n + 16];
    }

    const float aS0 = Anode[s * 32 + n];
    const float aS1 = Anode[s * 32 + n + 16];
    const float sx = nuclei[s * 3 + 0], sy = nuclei[s * 3 + 1], sz = nuclei[s * 3 + 2];

    float hacc0 = 0.0f, hacc1 = 0.0f;

    auto do_chunk = [&](const int cbase, const bool tail) {
        // Build this lane's A row: Bessel RBF of edge (cbase + n).
        const int eiA = cbase + n;
        int rA = eiA + (eiA >= s ? 1 : 0);
        if (tail) rA = (rA > NN - 1) ? (NN - 1) : rA;   // pad edge -> safe row
        const float dx = sx - nuclei[rA * 3 + 0];
        const float dy = sy - nuclei[rA * 3 + 1];
        const float dz = sz - nuclei[rA * 3 + 2];
        const float x  = sqrtf(dx * dx + dy * dy + dz * dz) + 1e-8f;
        const float sc = RBF_SCALE * __builtin_amdgcn_rcpf(x);
        v16h av;
        #pragma unroll
        for (int e = 0; e < 16; ++e) av[e] = (_Float16)(sc * __sinf(fk[e] * x));

        v8f c0 = {}, c1 = {};
        c0 = __builtin_amdgcn_wmma_f32_16x16x32_f16(false, av, false, wb0,
                                                    (short)0, c0, false, false);
        c1 = __builtin_amdgcn_wmma_f32_16x16x32_f16(false, av, false, wb1,
                                                    (short)0, c1, false, false);

        // pre-activation = C + A[s] + B[r]; silu; accumulate column sums
        #pragma unroll
        for (int g = 0; g < 8; ++g) {
            const int ei = cbase + g + kh * 8;
            int r = ei + (ei >= s ? 1 : 0);
            if (tail) r = (r > NN - 1) ? (NN - 1) : r;
            const float2 bv = Bpk[r * 16 + n];
            const float p0 = c0[g] + aS0 + bv.x;
            const float p1 = c1[g] + aS1 + bv.y;
            const float h0 = silu_f(p0);
            const float h1 = silu_f(p1);
            if (tail) {
                const bool valid = (ei < EDGES_PER);
                hacc0 += valid ? h0 : 0.0f;
                hacc1 += valid ? h1 : 0.0f;
            } else {
                hacc0 += h0;
                hacc1 += h1;
            }
        }
    };

    for (int chunk = wave; chunk < CHUNKS; chunk += 8) {
        if (chunk != CHUNKS - 1) do_chunk(chunk * 16, false);  // steady state
        else                     do_chunk(chunk * 16, true);   // pad-edge chunk
    }

    // lanes L and L+16 hold the same output columns -> fold, then cross-wave reduce
    hacc0 += __shfl_xor(hacc0, 16, 32);
    hacc1 += __shfl_xor(hacc1, 16, 32);
    if (lane < 16) { redbuf[wave][n] = hacc0; redbuf[wave][n + 16] = hacc1; }
    __syncthreads();
    if (tid < 32) {
        float hs = 0.0f;
        #pragma unroll
        for (int w = 0; w < 8; ++w) hs += redbuf[w][tid];
        hsumS[tid] = hs;
    }
    __syncthreads();
    if (tid < 32) {
        float o = 0.0f;
        #pragma unroll
        for (int k = 0; k < 32; ++k) o += hsumS[k] * W1[k * 32 + tid];
        msg[s * 32 + tid] = o * (1.0f / 767.0f) + b1[tid];
    }
}

// ---------------------------------------------------------------------------
// K3: node update layer 1 (one 64-thread block per node) + A2/B2 precompute.
// B2 written pair-interleaved for the layer-2 edge pass.
// ---------------------------------------------------------------------------
__global__ __launch_bounds__(64) void node_update1_kernel(
    const float* __restrict__ emb0, const float* __restrict__ msg1,
    const float* __restrict__ up1_w0, const float* __restrict__ up1_b0,
    const float* __restrict__ up1_w1, const float* __restrict__ up1_b1,
    const float* __restrict__ mp2_w0, const float* __restrict__ mp2_b0,
    float* __restrict__ emb1, float* __restrict__ A2, float* __restrict__ B2pk)
{
    __shared__ float uin[64], hid[64], e1[64];
    const int i = blockIdx.x, j = threadIdx.x;
    uin[j] = (j < 32) ? emb0[i * 32 + j] : msg1[i * 32 + (j - 32)];
    __syncthreads();
    float a = up1_b0[j];
    for (int k = 0; k < 64; ++k) a += uin[k] * up1_w0[k * 64 + j];
    hid[j] = silu_f(a);
    __syncthreads();
    float o = up1_b1[j];
    for (int k = 0; k < 64; ++k) o += hid[k] * up1_w1[k * 64 + j];
    emb1[i * 64 + j] = o;
    e1[j] = o;
    __syncthreads();
    if (j < 32) {
        float a2 = mp2_b0[j], b2 = 0.0f;
        for (int k = 0; k < 64; ++k) {
            a2 += e1[k] * mp2_w0[k * 32 + j];
            b2 += e1[k] * mp2_w0[(64 + k) * 32 + j];
        }
        A2[i * 32 + j] = a2;
        B2pk[i * 32 + ((j & 15) << 1) + (j >> 4)] = b2;
    }
}

// ---------------------------------------------------------------------------
// K5: node update layer 2 (+residual), node head, agg atomic accumulation.
// ---------------------------------------------------------------------------
__global__ __launch_bounds__(64) void node_update2_kernel(
    const int* __restrict__ charges,
    const float* __restrict__ emb0, const float* __restrict__ emb1,
    const float* __restrict__ msg2,
    const float* __restrict__ up2_w0, const float* __restrict__ up2_b0,
    const float* __restrict__ up2_w1, const float* __restrict__ up2_b1,
    const float* __restrict__ no_w0, const float* __restrict__ no_b0,
    const float* __restrict__ no_w1, const float* __restrict__ no_embed,
    float* __restrict__ aggsum, float* __restrict__ node_out)
{
    __shared__ float uin[96], hid[64], agg[160], h3[3];
    const int i = blockIdx.x, j = threadIdx.x;
    const float e1j = emb1[i * 64 + j];
    uin[j] = e1j;
    if (j < 32) uin[64 + j] = msg2[i * 32 + j];
    __syncthreads();
    float a = up2_b0[j];
    for (int k = 0; k < 96; ++k) a += uin[k] * up2_w0[k * 64 + j];
    hid[j] = silu_f(a);
    __syncthreads();
    float o = up2_b1[j];
    for (int k = 0; k < 64; ++k) o += hid[k] * up2_w1[k * 64 + j];
    const float e2j = e1j + o;                 // residual (64 == 64)
    agg[32 + j] = e1j;
    agg[96 + j] = e2j;
    float e0j = 0.0f;
    if (j < 32) { e0j = emb0[i * 32 + j]; agg[j] = e0j; }
    __syncthreads();
    if (j < 3) {
        float h = no_b0[j];
        for (int k = 0; k < 160; ++k) h += agg[k] * no_w0[k * 3 + j];
        h3[j] = silu_f(h);
    }
    __syncthreads();
    if (j < 3) {
        float v = no_embed[charges[i] * 3 + j];  // no final bias in node head
        for (int k = 0; k < 3; ++k) v += h3[k] * no_w1[k * 3 + j];
        node_out[i * 3 + j] = v;
    }
    atomicAdd(&aggsum[32 + j], e1j);
    atomicAdd(&aggsum[96 + j], e2j);
    if (j < 32) atomicAdd(&aggsum[j], e0j);
}

// ---------------------------------------------------------------------------
// K6: global head: silu(mean(agg) @ go_w0 + b0) * go_w1 + b1  -> d_out[2304]
// ---------------------------------------------------------------------------
__global__ __launch_bounds__(256) void global_out_kernel(
    const float* __restrict__ aggsum,
    const float* __restrict__ go_w0, const float* __restrict__ go_b0,
    const float* __restrict__ go_w1, const float* __restrict__ go_b1,
    float* __restrict__ out)
{
    __shared__ float red[256];
    const int t = threadIdx.x;
    red[t] = (t < 160) ? (aggsum[t] * (1.0f / 768.0f)) * go_w0[t] : 0.0f;
    __syncthreads();
    for (int st = 128; st > 0; st >>= 1) {
        if (t < st) red[t] += red[t + st];
        __syncthreads();
    }
    if (t == 0) {
        float h = silu_f(red[0] + go_b0[0]);
        out[0] = h * go_w1[0] + go_b1[0];
    }
}

// ---------------------------------------------------------------------------
extern "C" void kernel_launch(void* const* d_in, const int* in_sizes, int n_in,
                              void* d_out, int out_size, void* d_ws, size_t ws_size,
                              hipStream_t stream)
{
    const float* nuclei  = (const float*)d_in[0];
    const int*   charges = (const int*)  d_in[1];
    const float* fparam  = (const float*)d_in[2];
    const float* embT    = (const float*)d_in[3];
    const float* mp1_w0  = (const float*)d_in[4];
    const float* mp1_b0  = (const float*)d_in[5];
    const float* mp1_w1  = (const float*)d_in[6];
    const float* mp1_b1  = (const float*)d_in[7];
    const float* up1_w0  = (const float*)d_in[8];
    const float* up1_b0  = (const float*)d_in[9];
    const float* up1_w1  = (const float*)d_in[10];
    const float* up1_b1  = (const float*)d_in[11];
    const float* mp2_w0  = (const float*)d_in[12];
    const float* mp2_b0  = (const float*)d_in[13];
    const float* mp2_w1  = (const float*)d_in[14];
    const float* mp2_b1  = (const float*)d_in[15];
    const float* up2_w0  = (const float*)d_in[16];
    const float* up2_b0  = (const float*)d_in[17];
    const float* up2_w1  = (const float*)d_in[18];
    const float* up2_b1  = (const float*)d_in[19];
    const float* no_w0   = (const float*)d_in[20];
    const float* no_b0   = (const float*)d_in[21];
    const float* no_w1   = (const float*)d_in[22];
    const float* no_embed= (const float*)d_in[23];
    const float* go_w0   = (const float*)d_in[24];
    const float* go_b0   = (const float*)d_in[25];
    const float* go_w1   = (const float*)d_in[26];
    const float* go_b1   = (const float*)d_in[27];

    float* ws     = (float*)d_ws;
    float* emb0   = ws;                    // 768*32
    float* A1     = emb0 + NN * 32;        // 768*32
    float* B1pk   = A1   + NN * 32;        // 768*32 (pair-interleaved)
    float* msg1   = B1pk + NN * 32;        // 768*32
    float* emb1   = msg1 + NN * 32;        // 768*64
    float* A2     = emb1 + NN * 64;        // 768*32
    float* B2pk   = A2   + NN * 32;        // 768*32 (pair-interleaved)
    float* msg2   = B2pk + NN * 32;        // 768*32
    float* aggsum = msg2 + NN * 32;        // 160

    float* node_out = (float*)d_out;       // 768*3
    float* gout     = node_out + NN * 3;   // 1

    node_init_kernel<<<96, 256, 0, stream>>>(charges, embT, mp1_w0, mp1_b0,
                                             emb0, A1, B1pk, aggsum);
    edge_msg_kernel<<<NN, 256, 0, stream>>>(nuclei, fparam, A1, (const float2*)B1pk,
                                            mp1_w0 + 64 * 32, mp1_w1, mp1_b1, msg1);
    node_update1_kernel<<<NN, 64, 0, stream>>>(emb0, msg1, up1_w0, up1_b0,
                                               up1_w1, up1_b1, mp2_w0, mp2_b0,
                                               emb1, A2, B2pk);
    edge_msg_kernel<<<NN, 256, 0, stream>>>(nuclei, fparam, A2, (const float2*)B2pk,
                                            mp2_w0 + 128 * 32, mp2_w1, mp2_b1, msg2);
    node_update2_kernel<<<NN, 64, 0, stream>>>(charges, emb0, emb1, msg2,
                                               up2_w0, up2_b0, up2_w1, up2_b1,
                                               no_w0, no_b0, no_w1, no_embed,
                                               aggsum, node_out);
    global_out_kernel<<<1, 256, 0, stream>>>(aggsum, go_w0, go_b0, go_w1, go_b1, gout);
}
